// SDF_10754598109360
// MI455X (gfx1250) — matrix operand order repaired and checked by reading, not compiled, not run on hardware
//
#include <hip/hip_runtime.h>
#include <math.h>

typedef __fp16   pk2  __attribute__((ext_vector_type(2)));   // cvt_pkrtz result
typedef _Float16 h8   __attribute__((ext_vector_type(8)));
typedef _Float16 v16h __attribute__((ext_vector_type(16)));
typedef float    v8f  __attribute__((ext_vector_type(8)));

#define WAVES 4
#define LDH 72          // padded activation row stride in halves (144B = 9*16B)
#define NEAR_T 0.2f
#define FAR_T  2.0f
#define EPS_T  0.001f
#define MARCH_ITERS 192
#define NFRAG 10        // 8 x W1 (ks*4+nt) | 2 x W2[:,0:16] (ks)

// LDS ops from one wave complete in order (DScnt); force the compiler to issue
// pending stores / wait before dependent fragment loads, and keep it from
// hoisting LDS weight-fragment loads into long-lived registers.
__device__ __forceinline__ void wave_lds_fence() {
  asm volatile("s_wait_dscnt 0" ::: "memory");
}

// single-instruction relu: v_med3_f32(x, 0, +BIG)
__device__ __forceinline__ float relu1(float x) {
  return __builtin_amdgcn_fmed3f(x, 0.f, 3.0e38f);
}

// A-fragment (16x32 f16) per ISA 7.12.2: lane row = l16; h=0: K 0..7/16..23,
// h=1: K 8..15/24..31 -> two contiguous 16B chunks per lane.
__device__ __forceinline__ v16h load_afrag(const _Float16* buf, int ldh,
                                           int l16, int h, int k0) {
  const _Float16* p = buf + l16 * ldh + k0 + 8 * h;
  h8 lo = *(const h8*)(p);
  h8 hi = *(const h8*)(p + 16);
  v16h a;
#pragma unroll
  for (int i = 0; i < 8; ++i) { a[i] = lo[i]; a[8 + i] = hi[i]; }
  return a;
}

// B-fragment (32x16 f16): element i <-> K = k0 + 8h + i (i<8) / k0 + 16 + 8h + (i-8) (i>=8), N = n0 + l16,
// gathered from a row-major f32 weight matrix with zero padding.
// transp=true reads W^T (B[k][n] = W[n][k]).
__device__ __forceinline__ v16h load_bfrag(const float* W, int ldw, int krows,
                                           int ncols, int k0, int n0,
                                           bool transp, int l16, int h) {
  v16h b;
  int n = n0 + l16;
#pragma unroll
  for (int i = 0; i < 16; ++i) {
    int k = k0 + 8 * h + (i < 8 ? i : i + 8);
    float v = 0.f;
    if (k < krows && n < ncols) v = transp ? W[n * ldw + k] : W[k * ldw + n];
    b[i] = (_Float16)v;
  }
  return b;
}

__device__ __forceinline__ v8f wmma_f32(v16h a, v16h b, v8f c) {
  return __builtin_amdgcn_wmma_f32_16x16x32_f16(false, a, false, b, (short)0,
                                                c, false, false);
}

// D element i <-> (M = mt*16 + i + 8h, N = nt*16 + l16): relu (v_med3) +
// packed cvt (v_cvt_pk_f16_f32) + b16 / b16_d16_hi scatter-store pair.
__device__ __forceinline__ void store_relu(_Float16* out, int ldh, int l16,
                                           int h, int mt, int nt, v8f c) {
  _Float16* p = out + (mt * 16 + 8 * h) * ldh + nt * 16 + l16;
#pragma unroll
  for (int i = 0; i < 8; i += 2) {
    pk2 pk = __builtin_amdgcn_cvt_pkrtz(relu1(c[i]), relu1(c[i + 1]));
    p[(i)     * ldh] = (_Float16)pk[0];
    p[(i + 1) * ldh] = (_Float16)pk[1];
  }
}

// two-M-tile GEMM with B gathered from global f32 (cold path)
template <int KSTEPS, int NTILES, typename EmitF>
__device__ __forceinline__ void gemm2(const _Float16* inbuf, int ldin,
                                      const float* W, int ldw, int krows,
                                      int ncols, bool transp, const float* bias,
                                      int l16, int h, EmitF emit) {
#pragma unroll
  for (int mt = 0; mt < 2; ++mt) {
    v16h a[KSTEPS];
#pragma unroll
    for (int ks = 0; ks < KSTEPS; ++ks)
      a[ks] = load_afrag(inbuf + mt * 16 * ldin, ldin, l16, h, 32 * ks);
#pragma unroll
    for (int nt = 0; nt < NTILES; ++nt) {
      int n = nt * 16 + l16;
      float bv = (bias && n < ncols) ? bias[n] : 0.f;
      v8f c = {bv, bv, bv, bv, bv, bv, bv, bv};
#pragma unroll
      for (int ks = 0; ks < KSTEPS; ++ks) {
        v16h b = load_bfrag(W, ldw, krows, ncols, 32 * ks, nt * 16, transp, l16, h);
        c = wmma_f32(a[ks], b, c);
      }
      emit(mt, nt, c);
    }
  }
}

__global__ __launch_bounds__(32 * WAVES)
void sdf_march_render(const float* __restrict__ rays,
                      const float* __restrict__ W0, const float* __restrict__ b0,
                      const float* __restrict__ W1, const float* __restrict__ b1,
                      const float* __restrict__ W2, const float* __restrict__ b2,
                      const float* __restrict__ RW0, const float* __restrict__ Rb0,
                      const float* __restrict__ RW1, const float* __restrict__ Rb1,
                      float* __restrict__ out) {
  // per-wave activation scratch: 32 rays x 64 (+8 pad) f16, row-major
  __shared__ __align__(16) _Float16 s_h1[WAVES][32][LDH]; // h1 / G1
  __shared__ __align__(16) _Float16 s_h2[WAVES][32][LDH]; // h2
  __shared__ __align__(16) _Float16 s_f [WAVES][32][64];  // G2, then feat
  // pre-swizzled weight B-fragments, chunk-split for 2-way-max bank conflicts
  __shared__ __align__(16) _Float16 s_wb[NFRAG][2][32][8];
  __shared__ float s_d  [WAVES][32];
  __shared__ float s_hit[WAVES][32];

  const int tid  = threadIdx.x;
  const int w    = tid >> 5;
  const int lane = tid & 31;
  const int h    = lane >> 4;
  const int l16  = lane & 15;
  const int base = (blockIdx.x * WAVES + w) * 32; // 32 rays per wave, 1/lane

  // ---- cooperative one-time build of march-loop weight fragments ----
  for (int idx = tid; idx < NFRAG * 32; idx += 32 * WAVES) {
    int f = idx >> 5, l = idx & 31;
    int fl = l & 15, fh = l >> 4;
    v16h b;
    if (f < 8) { // W1: f = ks*4 + nt
      b = load_bfrag(W1, 64, 64, 64, (f >> 2) * 32, (f & 3) * 16, false, fl, fh);
    } else {     // W2 cols 0..15, ks = f-8
      b = load_bfrag(W2, 33, 64, 33, (f - 8) * 32, 0, false, fl, fh);
    }
    h8 lo, hi;
#pragma unroll
    for (int i = 0; i < 8; ++i) { lo[i] = b[i]; hi[i] = b[8 + i]; }
    *(h8*)&s_wb[f][0][l][0] = lo;
    *(h8*)&s_wb[f][1][l][0] = hi;
  }
  __syncthreads();

  auto bw = [&](int f) -> v16h {
    h8 lo = *(const h8*)&s_wb[f][0][lane][0];
    h8 hi = *(const h8*)&s_wb[f][1][lane][0];
    v16h b;
#pragma unroll
    for (int i = 0; i < 8; ++i) { b[i] = lo[i]; b[8 + i] = hi[i]; }
    return b;
  };

  // W0 B-fragments are mostly zero (K = 0..2): keep in registers.
  v16h w0b[4];
#pragma unroll
  for (int nt = 0; nt < 4; ++nt) {
    v16h b = {};
    _Float16 z = (_Float16)0.f;
    int n = nt * 16 + l16;
    b[0] = (h == 0) ? (_Float16)W0[0 * 64 + n] : z;
    b[1] = (h == 0) ? (_Float16)W0[1 * 64 + n] : z;
    b[2] = (h == 0) ? (_Float16)W0[2 * 64 + n] : z;
    w0b[nt] = b;
  }

  // each lane owns ray (base + lane)
  const float* rp = rays + (size_t)(base + lane) * 6;
  float ox = rp[0], oy = rp[1], oz = rp[2];
  float dx = rp[3], dy = rp[4], dz = rp[5];

  float b0r[4], b1r[4];
#pragma unroll
  for (int nt = 0; nt < 4; ++nt) {
    b0r[nt] = b0[nt * 16 + l16];
    b1r[nt] = b1[nt * 16 + l16];
  }
  float b2r = b2[l16];

  // build layer-0 A fragments for both M-tiles from per-lane p via shuffles
  auto build_a0 = [&](float px, float py, float pz, v16h a0[2]) {
    float sx[2], sy[2], sz[2];
    sx[0] = __shfl(px, l16);      sy[0] = __shfl(py, l16);      sz[0] = __shfl(pz, l16);
    sx[1] = __shfl(px, l16 + 16); sy[1] = __shfl(py, l16 + 16); sz[1] = __shfl(pz, l16 + 16);
    _Float16 z = (_Float16)0.f;
#pragma unroll
    for (int mt = 0; mt < 2; ++mt) {
      v16h a = {};
      a[0] = (h == 0) ? (_Float16)sx[mt] : z;
      a[1] = (h == 0) ? (_Float16)sy[mt] : z;
      a[2] = (h == 0) ? (_Float16)sz[mt] : z;
      a0[mt] = a;
    }
  };

  // ---------------- sphere march: 28 WMMA / iteration / 32 rays ----------------
  float t = NEAR_T;
  bool hit = false, rem = true;
  for (int it = 0; it < MARCH_ITERS; ++it) {
    if (__ballot(rem ? 1 : 0) == 0ull) break; // wave-uniform early exit

    float px = ox + dx * t, py = oy + dy * t, pz = oz + dz * t;
    v16h a0[2];
    build_a0(px, py, pz, a0);

    // layer0: h1 = relu(p @ W0 + b0)
#pragma unroll
    for (int nt = 0; nt < 4; ++nt) {
      float bv = b0r[nt];
#pragma unroll
      for (int mt = 0; mt < 2; ++mt) {
        v8f c = {bv, bv, bv, bv, bv, bv, bv, bv};
        c = wmma_f32(a0[mt], w0b[nt], c);
        store_relu(&s_h1[w][0][0], LDH, l16, h, mt, nt, c);
      }
    }
    wave_lds_fence();

    // layer1: h2 = relu(h1 @ W1 + b1)
    v16h a1[2][2];
#pragma unroll
    for (int mt = 0; mt < 2; ++mt)
#pragma unroll
      for (int ks = 0; ks < 2; ++ks)
        a1[mt][ks] = load_afrag(&s_h1[w][mt * 16][0], LDH, l16, h, 32 * ks);
#pragma unroll
    for (int nt = 0; nt < 4; ++nt) {
      v16h bk0 = bw(nt);      // W1 ks0
      v16h bk1 = bw(4 + nt);  // W1 ks1
      float bv = b1r[nt];
#pragma unroll
      for (int mt = 0; mt < 2; ++mt) {
        v8f c = {bv, bv, bv, bv, bv, bv, bv, bv};
        c = wmma_f32(a1[mt][0], bk0, c);
        c = wmma_f32(a1[mt][1], bk1, c);
        store_relu(&s_h2[w][0][0], LDH, l16, h, mt, nt, c);
      }
    }
    wave_lds_fence();

    // layer2 (only col 0 needed): d = h2 @ W2[:,0] + b2[0]
    {
      v16h bk0 = bw(8), bk1 = bw(9);
#pragma unroll
      for (int mt = 0; mt < 2; ++mt) {
        v16h ak0 = load_afrag(&s_h2[w][mt * 16][0], LDH, l16, h, 0);
        v16h ak1 = load_afrag(&s_h2[w][mt * 16][0], LDH, l16, h, 32);
        v8f c = {b2r, b2r, b2r, b2r, b2r, b2r, b2r, b2r};
        c = wmma_f32(ak0, bk0, c);
        c = wmma_f32(ak1, bk1, c);
        if (l16 == 0) { // N==0 column: lanes 0 (M=+0..7) and 16 (M=+8..15)
#pragma unroll
          for (int i = 0; i < 8; ++i) s_d[w][mt * 16 + i + 8 * h] = c[i];
        }
      }
    }
    wave_lds_fence();
    float dist = s_d[w][lane];

    bool newhit = rem && (dist < EPS_T) && (t <= FAR_T);
    hit = hit || newhit;
    t = rem ? t + dist : t;
    rem = rem && !hit && (t <= FAR_T);
  }

  // -------- shading: forward at pts, VJP normal, render MLP --------
  float px = ox + dx * t, py = oy + dy * t, pz = oz + dz * t;
  s_hit[w][lane] = hit ? 1.f : 0.f;

  // forward layer0 at pts
  {
    v16h a0[2];
    build_a0(px, py, pz, a0);
#pragma unroll
    for (int nt = 0; nt < 4; ++nt) {
      float bv = b0r[nt];
#pragma unroll
      for (int mt = 0; mt < 2; ++mt) {
        v8f c = {bv, bv, bv, bv, bv, bv, bv, bv};
        c = wmma_f32(a0[mt], w0b[nt], c);
        store_relu(&s_h1[w][0][0], LDH, l16, h, mt, nt, c);
      }
    }
  }
  wave_lds_fence();
  gemm2<2, 4>(&s_h1[w][0][0], LDH, W1, 64, 64, 64, false, b1, l16, h,
    [&](int mt, int nt, v8f c) { store_relu(&s_h2[w][0][0], LDH, l16, h, mt, nt, c); });
  wave_lds_fence();

  // G2 = (h2 > 0) ? W2[:,0] : 0   (one row per lane, vectorized)
#pragma unroll
  for (int kc = 0; kc < 8; ++kc) {
    h8 hv = *(const h8*)&s_h2[w][lane][kc * 8];
    h8 g;
#pragma unroll
    for (int j = 0; j < 8; ++j)
      g[j] = ((float)hv[j] > 0.f) ? (_Float16)W2[(kc * 8 + j) * 33] : (_Float16)0.f;
    *(h8*)&s_f[w][lane][kc * 8] = g;
  }
  wave_lds_fence();

  // G1 = (G2 @ W1^T) * (h1 > 0)   (overwrites s_h1 in place)
  gemm2<2, 4>(&s_f[w][0][0], 64, W1, 64, 64, 64, true, nullptr, l16, h,
    [&](int mt, int nt, v8f c) {
#pragma unroll
      for (int i = 0; i < 8; ++i) {
        int m = mt * 16 + i + 8 * h, n = nt * 16 + l16;
        bool on = (float)s_h1[w][m][n] > 0.f;
        s_h1[w][m][n] = (_Float16)(on ? c[i] : 0.f);
      }
    });
  wave_lds_fence();

  // s_f dead (G2 consumed): zero it for feat, drop in pts/r_d (own row)
  {
    h8 z = {};
#pragma unroll
    for (int kc = 0; kc < 8; ++kc) *(h8*)&s_f[w][lane][kc * 8] = z;
  }
  s_f[w][lane][0] = (_Float16)px;
  s_f[w][lane][1] = (_Float16)py;
  s_f[w][lane][2] = (_Float16)pz;
  s_f[w][lane][3] = (_Float16)dx;
  s_f[w][lane][4] = (_Float16)dy;
  s_f[w][lane][5] = (_Float16)dz;
  wave_lds_fence();

  // nrm = G1 @ W0^T (64x3), masked by hit -> feat cols 6..8
  gemm2<2, 1>(&s_h1[w][0][0], LDH, W0, 64, 64, 3, true, nullptr, l16, h,
    [&](int mt, int nt, v8f c) {
      (void)nt;
      if (l16 < 3) {
#pragma unroll
        for (int i = 0; i < 8; ++i) {
          int m = mt * 16 + i + 8 * h;
          s_f[w][m][6 + l16] = (_Float16)(s_hit[w][m] != 0.f ? c[i] : 0.f);
        }
      }
    });

  // latent = (h2 @ W2 + b2)[:, 1:33] -> feat cols 9..40
  gemm2<2, 3>(&s_h2[w][0][0], LDH, W2, 33, 64, 33, false, b2, l16, h,
    [&](int mt, int nt, v8f c) {
      int n = nt * 16 + l16;
      if (n >= 1 && n <= 32) {
#pragma unroll
        for (int i = 0; i < 8; ++i)
          s_f[w][mt * 16 + i + 8 * h][8 + n] = (_Float16)c[i];
      }
    });
  wave_lds_fence();

  // render layer 0: relu(feat @ RW0 + Rb0) -> s_h1 (G1 dead)
  gemm2<2, 4>(&s_f[w][0][0], 64, RW0, 64, 41, 64, false, Rb0, l16, h,
    [&](int mt, int nt, v8f c) { store_relu(&s_h1[w][0][0], LDH, l16, h, mt, nt, c); });
  wave_lds_fence();

  // render layer 1: rgb = sigmoid(hr @ RW1 + Rb1), masked by hit
  gemm2<2, 1>(&s_h1[w][0][0], LDH, RW1, 3, 64, 3, false, Rb1, l16, h,
    [&](int mt, int nt, v8f c) {
      (void)nt;
      if (l16 < 3) {
#pragma unroll
        for (int i = 0; i < 8; ++i) {
          int m = mt * 16 + i + 8 * h;
          float v = 1.f / (1.f + __expf(-c[i]));
          out[(size_t)(base + m) * 3 + l16] = (s_hit[w][m] != 0.f) ? v : 0.f;
        }
      }
    });
}

extern "C" void kernel_launch(void* const* d_in, const int* in_sizes, int n_in,
                              void* d_out, int out_size, void* d_ws,
                              size_t ws_size, hipStream_t stream) {
  (void)n_in; (void)out_size; (void)d_ws; (void)ws_size;
  const float* rays = (const float*)d_in[0];
  const float* W0   = (const float*)d_in[1];
  const float* b0   = (const float*)d_in[2];
  const float* W1   = (const float*)d_in[3];
  const float* b1   = (const float*)d_in[4];
  const float* W2   = (const float*)d_in[5];
  const float* b2   = (const float*)d_in[6];
  const float* RW0  = (const float*)d_in[7];
  const float* Rb0  = (const float*)d_in[8];
  const float* RW1  = (const float*)d_in[9];
  const float* Rb1  = (const float*)d_in[10];
  float* out = (float*)d_out;

  int n_rays = in_sizes[0] / 6;                   // 262144
  int tiles  = n_rays / 32;                       // 8192 (32 rays per wave)
  int blocks = (tiles + WAVES - 1) / WAVES;       // 2048 blocks of 4 waves

  sdf_march_render<<<blocks, 32 * WAVES, 0, stream>>>(
      rays, W0, b0, W1, b1, W2, b2, RW0, Rb0, RW1, Rb1, out);
}
